// PolicyNetwork_59141699666038
// MI455X (gfx1250) — matrix operand order repaired and checked by reading, not compiled, not run on hardware
//
#include <hip/hip_runtime.h>

// Problem constants (match reference)
#define B_TOTAL  32768
#define N_HW     32
#define TASK_DIM 17
#define HW_DIM   16
#define EMB      64

// Tiling
#define TB       8                  // batch rows per workgroup
#define R_ROWS   (TB * N_HW)        // 256 hw rows per workgroup
#define N_TILES  (R_ROWS / 16)      // 16 WMMA row-tiles
#define WAVES    8
#define TPB      256

// LDS strides in f16 elems — all multiples of 8 (16B) so v8h loads are aligned,
// and not multiples of 32 to stagger bank groups (72*2=144B rows).
#define HW_STRIDE 72
#define TE_STRIDE 72
#define WB_STRIDE 72
#define VH_STRIDE 136

// Packed-fragment weight offsets in d_ws / s_wpack (f16 units).
// Fragment = 512 f16 (32 lanes x 16), f = kc*NT + nt.
#define TE1P_OFF 0       // 4 frags (NT=4, KC=1)
#define TE2P_OFF 2048    // 8 frags (NT=4, KC=2)
#define HE1P_OFF 6144    // 4 frags
#define HE2P_OFF 8192    // 8 frags
#define SC1P_OFF 12288   // 16 frags (NT=4, KC=4)
#define SC2P_OFF 20480   // 4 frags (NT=2, KC=2)
#define RJ1P_OFF 22528   // 4 frags (NT=2, KC=2)
#define VH1P_OFF 24576   // 16 frags
#define WPACK_TOTAL 32768   // f16 elems = 64 KB

typedef __attribute__((ext_vector_type(16))) _Float16 v16h;
typedef __attribute__((ext_vector_type(8)))  _Float16 v8h;
typedef __attribute__((ext_vector_type(8)))  float    v8f;

// Zero-instruction compiler fence: LDS is architecturally in-order per wave
// (CDNA5 ISA 7.1) — we only need to stop compiler reordering across the
// producer-store / consumer-load boundary.
#define LDS_CFENCE() asm volatile("" ::: "memory")

static __device__ __forceinline__ v8f wmma_f16(v16h a, v16h b, v8f c) {
  return __builtin_amdgcn_wmma_f32_16x16x32_f16(
      false, a, false, b, (short)0, c, false, false);
}

// B fragment (32x16 f16) pre-packed: one v16h load = 2x ds_load_b128
static __device__ __forceinline__ v16h lds_frag(const _Float16* base, int f, int lane) {
  return *(const v16h*)(base + f * 512 + lane * 16);
}

// A matrix (16x32 f16, MxK), ISA 7.12.2 layout: two contiguous 8-f16 runs per lane
static __device__ __forceinline__ v16h lds_load_a(const _Float16* p, int stride, int lane) {
  const int m  = lane & 15;
  const int hi = (lane >> 4) & 1;
  const _Float16* r = p + m * stride + hi * 8;
  v8h lo = *(const v8h*)r;
  v8h hh = *(const v8h*)(r + 16);
  return __builtin_shufflevector(lo, hh, 0,1,2,3,4,5,6,7,8,9,10,11,12,13,14,15);
}

// Same, but every row M broadcasts the SAME source row (task_emb in scorer)
static __device__ __forceinline__ v16h lds_load_a_row(const _Float16* rowp, int lane) {
  const int hi = (lane >> 4) & 1;
  const _Float16* r = rowp + hi * 8;
  v8h lo = *(const v8h*)r;
  v8h hh = *(const v8h*)(r + 16);
  return __builtin_shufflevector(lo, hh, 0,1,2,3,4,5,6,7,8,9,10,11,12,13,14,15);
}

// C/D (16x16 f32): VGPR v holds row v + 8*(lane>=16), col lane%16
static __device__ __forceinline__ void store_c_relu_f16(_Float16* dst, int stride, int ncol0,
                                                        const float* bias, v8f c, int lane) {
  const int n  = lane & 15;
  const int mb = ((lane >> 4) & 1) * 8;
  const float bv = bias[ncol0 + n];
#pragma unroll
  for (int v = 0; v < 8; ++v) {
    float x = fmaxf(c[v] + bv, 0.f);
    dst[(mb + v) * stride + ncol0 + n] = (_Float16)x;
  }
}

static __device__ __forceinline__ void store_c_relu_f32(float* dst, int stride, int ncol0,
                                                        const float* bias, v8f c, int lane) {
  const int n  = lane & 15;
  const int mb = ((lane >> 4) & 1) * 8;
  const float bv = bias[ncol0 + n];
#pragma unroll
  for (int v = 0; v < 8; ++v) {
    dst[(mb + v) * stride + ncol0 + n] = fmaxf(c[v] + bv, 0.f);
  }
}

// Layer-1 A tiles straight from global f32 (K zero-padded to 32)
static __device__ __forceinline__ v16h load_a_task_g(const float* task, int b0, int lane) {
  const int m   = lane & 15;
  const int row = b0 + (m < TB ? m : TB - 1);   // clamp padded rows
  const float* r = task + (size_t)row * TASK_DIM;
  const int hi = (lane >> 4) & 1;
  v16h a;
#pragma unroll
  for (int v = 0; v < 8; ++v) {
    const int k0 = ((v >> 2) << 4) + ((v & 3) << 1) + hi * 8;
    a[2 * v]     = (k0     < TASK_DIM) ? (_Float16)r[k0]     : (_Float16)0.f;
    a[2 * v + 1] = (k0 + 1 < TASK_DIM) ? (_Float16)r[k0 + 1] : (_Float16)0.f;
  }
  return a;
}

static __device__ __forceinline__ v16h load_a_hw_g(const float* hw, int r0, int lane) {
  const int m  = lane & 15;
  const int hi = (lane >> 4) & 1;
  const float4* r = (const float4*)(hw + (size_t)(r0 + m) * HW_DIM + hi * 8);
  const float4 x = r[0];
  const float4 y = r[1];
  v16h a = {};                       // K 16..31 zero pad
  a[0] = (_Float16)x.x; a[1] = (_Float16)x.y; a[2] = (_Float16)x.z; a[3] = (_Float16)x.w;
  a[4] = (_Float16)y.x; a[5] = (_Float16)y.y; a[6] = (_Float16)y.z; a[7] = (_Float16)y.w;
  return a;
}

static __device__ __forceinline__ void load_f(const float* g, float* lds, int n, int tid) {
  for (int i = tid; i < n; i += TPB) lds[i] = g[i];
}

// ---------------------------------------------------------------------------
// Pre-pack kernel: f32 weights -> f16 WMMA-B fragments in workspace
// ---------------------------------------------------------------------------
struct PackParams {
  const float* s0; const float* s1; const float* s2; const float* s3;
  const float* s4; const float* s5; const float* s6; const float* s7;
  _Float16* dst;
};

__global__ void pack_weights(PackParams P) {
  const float* srcs[8] = {P.s0, P.s1, P.s2, P.s3, P.s4, P.s5, P.s6, P.s7};
  const int K[8]   = {17, 64, 16, 64, 128, 64, 64, 128};
  const int N[8]   = {64, 64, 64, 64, 64, 32, 32, 64};
  const int NT[8]  = {4, 4, 4, 4, 4, 2, 2, 4};
  const int OFF[8] = {TE1P_OFF, TE2P_OFF, HE1P_OFF, HE2P_OFF,
                      SC1P_OFF, SC2P_OFF, RJ1P_OFF, VH1P_OFF};
  const int CNT[8] = {2048, 4096, 2048, 4096, 8192, 2048, 2048, 8192};
  const int gid = blockIdx.x * blockDim.x + threadIdx.x;
  const int gsz = gridDim.x * blockDim.x;
  for (int j = 0; j < 8; ++j) {
    const float* g = srcs[j];
    _Float16* d = P.dst + OFF[j];
    for (int i = gid; i < CNT[j]; i += gsz) {
      const int f    = i >> 9;
      const int lane = (i >> 4) & 31;
      const int e    = i & 15;
      const int kc   = f / NT[j];
      const int nt   = f - kc * NT[j];
      const int k    = kc * 32 + ((lane >> 4) & 1) * 8 + (e < 8 ? e : e + 8);
      const int n    = nt * 16 + (lane & 15);
      d[i] = (_Float16)((k < K[j]) ? g[k * N[j] + n] : 0.f);
    }
  }
}

// ---------------------------------------------------------------------------
// Fused policy-network kernel
// ---------------------------------------------------------------------------
struct KParams {
  const float* task; const float* hw; const unsigned char* mask;
  const float* te_b1; const float* te_b2; const float* he_b1; const float* he_b2;
  const float* sc_b1; const float* sc_b2; const float* sc_w3; const float* sc_b3;
  const float* rj_b1; const float* rj_w2; const float* rj_b2;
  const float* vh_b1; const float* vh_w2; const float* vh_b2;
  const _Float16* wpack;
  float* out;
};

__launch_bounds__(TPB)
__global__ void policy_fused_wmma(KParams P) {
  const int tid  = threadIdx.x;
  const int lane = tid & 31;
  const int wave = tid >> 5;
  const int b0   = blockIdx.x * TB;

  // ---------------- LDS (~146 KB -> 2 WGs per 320 KB WGP) ----------------
  __shared__ __align__(32) _Float16 s_wpack[WPACK_TOTAL];        // 64 KB
  __shared__ __align__(32) _Float16 s_task_emb[16 * TE_STRIDE];
  __shared__ __align__(32) _Float16 s_hw_emb[R_ROWS * HW_STRIDE];
  __shared__ __align__(32) _Float16 wbuf[WAVES][16 * WB_STRIDE];
  __shared__ __align__(16) float    h2buf[WAVES][16 * 36];
  __shared__ __align__(32) _Float16 s_vh_in[16 * VH_STRIDE];
  __shared__ float s_hw_scores[R_ROWS];
  __shared__ float s_rejects[TB];
  __shared__ float s_values[16];
  __shared__ __align__(16) float b_te1[64], b_te2[64], b_he1[64], b_he2[64];
  __shared__ __align__(16) float b_sc1[64], b_sc2[32], b_rj1[32], b_vh1[64];
  __shared__ __align__(16) float sw3[32], rw2[32], vw2[64];

  const _Float16* w_te1p = s_wpack + TE1P_OFF;
  const _Float16* w_te2p = s_wpack + TE2P_OFF;
  const _Float16* w_he1p = s_wpack + HE1P_OFF;
  const _Float16* w_he2p = s_wpack + HE2P_OFF;
  const _Float16* w_sc1p = s_wpack + SC1P_OFF;
  const _Float16* w_sc2p = s_wpack + SC2P_OFF;
  const _Float16* w_rj1p = s_wpack + RJ1P_OFF;
  const _Float16* w_vh1p = s_wpack + VH1P_OFF;

  // ---------------- Phase 0: async DMA packed weights global->LDS ----------------
  // GLOBAL_LOAD_ASYNC_TO_LDS_B128 (GVS mode): one instruction per 16B, no VGPR
  // round-trip; tracked by ASYNCcnt. LDS dest addr = low 32 bits of flat ptr.
  {
    const unsigned lds_base = (unsigned)(unsigned long long)(void*)&s_wpack[0];
    const void* gsrc = (const void*)P.wpack;
    for (int i = tid; i < (WPACK_TOTAL * 2) / 16; i += TPB) {
      const unsigned lds_addr = lds_base + (unsigned)(i * 16);
      const unsigned goff     = (unsigned)(i * 16);
      asm volatile("global_load_async_to_lds_b128 %0, %1, %2"
                   :: "v"(lds_addr), "v"(goff), "s"(gsrc)
                   : "memory");
    }
  }
  load_f(P.te_b1, b_te1, 64, tid);  load_f(P.te_b2, b_te2, 64, tid);
  load_f(P.he_b1, b_he1, 64, tid);  load_f(P.he_b2, b_he2, 64, tid);
  load_f(P.sc_b1, b_sc1, 64, tid);  load_f(P.sc_b2, b_sc2, 32, tid);
  load_f(P.rj_b1, b_rj1, 32, tid);  load_f(P.vh_b1, b_vh1, 64, tid);
  load_f(P.sc_w3, sw3, 32, tid);    load_f(P.rj_w2, rw2, 32, tid);
  load_f(P.vh_w2, vw2, 64, tid);
  asm volatile("s_wait_asynccnt 0" ::: "memory");   // our async DMAs done
  __syncthreads();

  // ---------------- Phase 1: task encoder (waves 0-3, one N-tile each) ----------------
  if (wave < 4) {
    v16h a = load_a_task_g(P.task, b0, lane);
    v8f  c = {};
    c = wmma_f16(a, lds_frag(w_te1p, wave, lane), c);
    store_c_relu_f16(&wbuf[0][0], WB_STRIDE, wave * 16, b_te1, c, lane);
  }
  __syncthreads();
  if (wave < 4) {
    v16h a0 = lds_load_a(&wbuf[0][0], WB_STRIDE, lane);
    v16h a1 = lds_load_a(&wbuf[0][0] + 32, WB_STRIDE, lane);
    v8f  c  = {};
    c = wmma_f16(a0, lds_frag(w_te2p, 0 * 4 + wave, lane), c);
    c = wmma_f16(a1, lds_frag(w_te2p, 1 * 4 + wave, lane), c);
    store_c_relu_f16(s_task_emb, TE_STRIDE, wave * 16, b_te2, c, lane);
  }
  __syncthreads();

  // ---------------- Phase 2: hw encoder (16 tiles, 2 per wave) ----------------
#pragma unroll
  for (int i = 0; i < N_TILES / WAVES; ++i) {
    const int t  = wave * (N_TILES / WAVES) + i;
    const int r0 = b0 * N_HW + t * 16;
    v16h a = load_a_hw_g(P.hw, r0, lane);
#pragma unroll
    for (int nt = 0; nt < 4; ++nt) {
      v8f c = {};
      c = wmma_f16(a, lds_frag(w_he1p, nt, lane), c);
      store_c_relu_f16(&wbuf[wave][0], WB_STRIDE, nt * 16, b_he1, c, lane);
    }
    LDS_CFENCE();
    v16h a0 = lds_load_a(&wbuf[wave][0], WB_STRIDE, lane);
    v16h a1 = lds_load_a(&wbuf[wave][0] + 32, WB_STRIDE, lane);
#pragma unroll
    for (int nt = 0; nt < 4; ++nt) {
      v8f c = {};
      c = wmma_f16(a0, lds_frag(w_he2p, 0 * 4 + nt, lane), c);
      c = wmma_f16(a1, lds_frag(w_he2p, 1 * 4 + nt, lane), c);
      store_c_relu_f16(s_hw_emb + t * 16 * HW_STRIDE, HW_STRIDE, nt * 16, b_he2, c, lane);
    }
  }
  __syncthreads();

  // ---------------- Phase 3: scorer (A = [task_emb | hw_emb], K=128) ----------------
  const float sc_b3v = P.sc_b3[0];
#pragma unroll
  for (int i = 0; i < N_TILES / WAVES; ++i) {
    const int t     = wave * (N_TILES / WAVES) + i;
    const int b_loc = t >> 1;
    const _Float16* trow = s_task_emb + b_loc * TE_STRIDE;
    v16h a0 = lds_load_a_row(trow, lane);
    v16h a1 = lds_load_a_row(trow + 32, lane);
    v16h a2 = lds_load_a(s_hw_emb + t * 16 * HW_STRIDE, HW_STRIDE, lane);
    v16h a3 = lds_load_a(s_hw_emb + t * 16 * HW_STRIDE + 32, HW_STRIDE, lane);
#pragma unroll
    for (int nt = 0; nt < 4; ++nt) {
      v8f c = {};
      c = wmma_f16(a0, lds_frag(w_sc1p, 0 * 4 + nt, lane), c);
      c = wmma_f16(a1, lds_frag(w_sc1p, 1 * 4 + nt, lane), c);
      c = wmma_f16(a2, lds_frag(w_sc1p, 2 * 4 + nt, lane), c);
      c = wmma_f16(a3, lds_frag(w_sc1p, 3 * 4 + nt, lane), c);
      store_c_relu_f16(&wbuf[wave][0], WB_STRIDE, nt * 16, b_sc1, c, lane);
    }
    LDS_CFENCE();
    v16h h0 = lds_load_a(&wbuf[wave][0], WB_STRIDE, lane);
    v16h h1 = lds_load_a(&wbuf[wave][0] + 32, WB_STRIDE, lane);
#pragma unroll
    for (int nt = 0; nt < 2; ++nt) {
      v8f c = {};
      c = wmma_f16(h0, lds_frag(w_sc2p, 0 * 2 + nt, lane), c);
      c = wmma_f16(h1, lds_frag(w_sc2p, 1 * 2 + nt, lane), c);
      store_c_relu_f32(&h2buf[wave][0], 36, nt * 16, b_sc2, c, lane);
    }
    LDS_CFENCE();
    if (lane < 16) {                               // final 32->1 dot per row
      float s = sc_b3v;
      const float4* hv = (const float4*)(&h2buf[wave][0] + lane * 36);
      const float4* wv = (const float4*)sw3;
#pragma unroll
      for (int q = 0; q < 8; ++q) {
        const float4 h = hv[q];
        const float4 w = wv[q];
        s += h.x * w.x + h.y * w.y + h.z * w.z + h.w * w.w;
      }
      s_hw_scores[t * 16 + lane] = s;
    }
  }
  __syncthreads();

  // ---------------- Phase 4a: mean(hw_emb) + build value-head input ----------------
  for (int i = tid; i < 16 * 8; i += TPB) {        // (b, k-group-of-8)
    const int b  = i >> 3;
    const int kg = i & 7;
    float acc[8] = {0.f, 0.f, 0.f, 0.f, 0.f, 0.f, 0.f, 0.f};
    v8h tv = {};
    if (b < TB) {
      const _Float16* base = s_hw_emb + (b * N_HW) * HW_STRIDE + kg * 8;
#pragma unroll 4
      for (int n = 0; n < N_HW; ++n) {
        const v8h h = *(const v8h*)(base + n * HW_STRIDE);
#pragma unroll
        for (int e = 0; e < 8; ++e) acc[e] += (float)h[e];
      }
      tv = *(const v8h*)(s_task_emb + b * TE_STRIDE + kg * 8);
    }
    v8h mv;
#pragma unroll
    for (int e = 0; e < 8; ++e) mv[e] = (_Float16)(acc[e] * (1.f / 32.f));
    *(v8h*)(s_vh_in + b * VH_STRIDE + kg * 8) = tv;
    *(v8h*)(s_vh_in + b * VH_STRIDE + 64 + kg * 8) = mv;
  }
  __syncthreads();

  // ---------------- Phase 4b: value head (wave 0) || reject head (wave 1, WMMA) ----------------
  if (wave == 0) {
    v16h a0 = lds_load_a(s_vh_in,      VH_STRIDE, lane);
    v16h a1 = lds_load_a(s_vh_in + 32, VH_STRIDE, lane);
    v16h a2 = lds_load_a(s_vh_in + 64, VH_STRIDE, lane);
    v16h a3 = lds_load_a(s_vh_in + 96, VH_STRIDE, lane);
#pragma unroll
    for (int nt = 0; nt < 4; ++nt) {
      v8f c = {};
      c = wmma_f16(a0, lds_frag(w_vh1p, 0 * 4 + nt, lane), c);
      c = wmma_f16(a1, lds_frag(w_vh1p, 1 * 4 + nt, lane), c);
      c = wmma_f16(a2, lds_frag(w_vh1p, 2 * 4 + nt, lane), c);
      c = wmma_f16(a3, lds_frag(w_vh1p, 3 * 4 + nt, lane), c);
      store_c_relu_f16(&wbuf[0][0], WB_STRIDE, nt * 16, b_vh1, c, lane);
    }
    LDS_CFENCE();
    if (lane < TB) {
      float s = P.vh_b2[0];
      const v8h* hv = (const v8h*)(&wbuf[0][0] + lane * WB_STRIDE);
#pragma unroll
      for (int q = 0; q < 8; ++q) {
        const v8h h = hv[q];
#pragma unroll
        for (int e = 0; e < 8; ++e) s += (float)h[e] * vw2[q * 8 + e];
      }
      s_values[lane] = s;
    }
  } else if (wave == 1) {
    // reject head: relu(task_emb @ rj_w1 + b) @ rj_w2 + b2 via WMMA
    v16h a0 = lds_load_a(s_task_emb, TE_STRIDE, lane);
    v16h a1 = lds_load_a(s_task_emb + 32, TE_STRIDE, lane);
#pragma unroll
    for (int nt = 0; nt < 2; ++nt) {
      v8f c = {};
      c = wmma_f16(a0, lds_frag(w_rj1p, 0 * 2 + nt, lane), c);
      c = wmma_f16(a1, lds_frag(w_rj1p, 1 * 2 + nt, lane), c);
      store_c_relu_f32(&h2buf[1][0], 36, nt * 16, b_rj1, c, lane);
    }
    LDS_CFENCE();
    if (lane < TB) {
      float s = P.rj_b2[0];
      const float4* hv = (const float4*)(&h2buf[1][0] + lane * 36);
      const float4* wv = (const float4*)rw2;
#pragma unroll
      for (int q = 0; q < 8; ++q) {
        const float4 h = hv[q];
        const float4 w = wv[q];
        s += h.x * w.x + h.y * w.y + h.z * w.z + h.w * w.w;
      }
      s_rejects[lane] = s;
    }
  }
  __syncthreads();

  // ---------------- Phase 5: mask + softmax(33) + outputs ----------------
  if (tid < TB) {
    const int b  = tid;
    const int bg = b0 + b;
    float sc[33];
    for (int n = 0; n < N_HW; ++n) {
      const unsigned char vm = P.mask[(size_t)bg * N_HW + n];
      sc[n] = vm ? s_hw_scores[b * N_HW + n] : -__builtin_inff();
    }
    sc[32] = s_rejects[b];
    float mx = sc[32];
    for (int j = 0; j < 32; ++j) mx = fmaxf(mx, sc[j]);
    float ex[33];
    float sum = 0.f;
    for (int j = 0; j < 33; ++j) { const float e = __expf(sc[j] - mx); ex[j] = e; sum += e; }
    const float inv = 1.f / sum;
    float* out = P.out;
    const size_t p_base = (size_t)bg * 33;
    const size_t s_base = (size_t)B_TOTAL * 33 + (size_t)B_TOTAL + (size_t)bg * 33;
    for (int j = 0; j < 33; ++j) {
      out[p_base + j] = ex[j] * inv;      // action_probs
      out[s_base + j] = sc[j];            // all_scores (masked = -inf)
    }
    out[(size_t)B_TOTAL * 33 + bg] = s_values[b];   // value
  }
}

extern "C" void kernel_launch(void* const* d_in, const int* in_sizes, int n_in,
                              void* d_out, int out_size, void* d_ws, size_t ws_size,
                              hipStream_t stream) {
  (void)in_sizes; (void)n_in; (void)out_size; (void)ws_size;

  PackParams PP;
  PP.s0 = (const float*)d_in[3];    // te_w1
  PP.s1 = (const float*)d_in[5];    // te_w2
  PP.s2 = (const float*)d_in[7];    // he_w1
  PP.s3 = (const float*)d_in[9];    // he_w2
  PP.s4 = (const float*)d_in[11];   // sc_w1
  PP.s5 = (const float*)d_in[13];   // sc_w2
  PP.s6 = (const float*)d_in[17];   // rj_w1
  PP.s7 = (const float*)d_in[21];   // vh_w1
  PP.dst = (_Float16*)d_ws;
  pack_weights<<<dim3(64), dim3(TPB), 0, stream>>>(PP);

  KParams P;
  P.task  = (const float*)d_in[0];
  P.hw    = (const float*)d_in[1];
  P.mask  = (const unsigned char*)d_in[2];
  P.te_b1 = (const float*)d_in[4];
  P.te_b2 = (const float*)d_in[6];
  P.he_b1 = (const float*)d_in[8];
  P.he_b2 = (const float*)d_in[10];
  P.sc_b1 = (const float*)d_in[12];
  P.sc_b2 = (const float*)d_in[14];
  P.sc_w3 = (const float*)d_in[15];
  P.sc_b3 = (const float*)d_in[16];
  P.rj_b1 = (const float*)d_in[18];
  P.rj_w2 = (const float*)d_in[19];
  P.rj_b2 = (const float*)d_in[20];
  P.vh_b1 = (const float*)d_in[22];
  P.vh_w2 = (const float*)d_in[23];
  P.vh_b2 = (const float*)d_in[24];
  P.wpack = (const _Float16*)d_ws;
  P.out   = (float*)d_out;
  policy_fused_wmma<<<dim3(B_TOTAL / TB), dim3(TPB), 0, stream>>>(P);
}
